// CategorySpecificLinear_82592221102548
// MI455X (gfx1250) — matrix operand order repaired and checked by reading, not compiled, not run on hardware
//
#include <hip/hip_runtime.h>

// ---------------------------------------------------------------------------
// Category-specific linear: out[b,s,h] = sum_i x[b,s,i] * W[cat[b],i,h] + bias
// bf16 WMMA (v_wmma_f32_16x16x32_bf16) with f32 accumulation.
// ---------------------------------------------------------------------------

typedef __attribute__((ext_vector_type(16))) __bf16 v16bf;
typedef __attribute__((ext_vector_type(8)))  __bf16 v8bf;
typedef __attribute__((ext_vector_type(8)))  float  v8f;

#define BATCH      64
#define SEQ        512
#define IN_DIM     1024
#define HID_DIM    4096

#define BM 128          // output rows per workgroup (S dim)
#define BN 128          // output cols per workgroup (H dim)
#define BK 32           // K step (bf16 WMMA K)
#define KSTR (BK + 8)   // padded LDS row stride (elements) to spread banks

__global__ __launch_bounds__(256)
void cat_linear_wmma_kernel(const float* __restrict__ x,
                            const int*   __restrict__ cat_ids,
                            const float* __restrict__ W,
                            const float* __restrict__ bias,
                            float*       __restrict__ out)
{
    __shared__ __bf16 As[BM * KSTR];   // X tile, [m][k], bf16
    __shared__ __bf16 Bs[BN * KSTR];   // W tile transposed, [n][k], bf16

    const int tid   = threadIdx.x;
    const int lane  = tid & 31;
    const int wave  = tid >> 5;        // 0..7
    const int waveM = wave & 1;        // 2 waves along M
    const int waveN = wave >> 1;       // 4 waves along N

    const int b  = blockIdx.z;
    const int gM = blockIdx.y * BM;    // row offset within S
    const int gN = blockIdx.x * BN;    // col offset within H
    const int cat = cat_ids[b];

    const float* Xb = x + ((size_t)b * SEQ + gM) * IN_DIM;
    const float* Wc = W + (size_t)cat * IN_DIM * HID_DIM + gN;

    // accumulators: 4 M-subtiles x 2 N-subtiles of 16x16 f32
    v8f acc[4][2];
#pragma unroll
    for (int mt = 0; mt < 4; ++mt)
#pragma unroll
        for (int nt = 0; nt < 2; ++nt)
#pragma unroll
            for (int v = 0; v < 8; ++v)
                acc[mt][nt][v] = 0.0f;

    // staging thread mapping (X): 128 rows x 32 cols, 16 floats/thread
    const int xr = tid >> 2;           // 0..63 (two row passes)
    const int xc = (tid & 3) * 8;      // 0,8,16,24

    // staging thread mapping (W): 32(k) x 128(n) -> Bs[n][k], 16 floats/thread
    const int wn = tid & 127;          // 0..127 (column n)
    const int wk = (tid >> 7) * 16;    // 0 or 16 (k base)

    for (int k0 = 0; k0 < IN_DIM; k0 += BK) {
        // ---- stage X tile (row-major [m][k]) ----
#pragma unroll
        for (int rr = 0; rr < 2; ++rr) {
            const int r = xr + rr * 64;
            const float* src = Xb + (size_t)r * IN_DIM + k0 + xc;
            float4 f0 = *(const float4*)(src);
            float4 f1 = *(const float4*)(src + 4);
            v8bf h;
            h[0] = (__bf16)f0.x; h[1] = (__bf16)f0.y;
            h[2] = (__bf16)f0.z; h[3] = (__bf16)f0.w;
            h[4] = (__bf16)f1.x; h[5] = (__bf16)f1.y;
            h[6] = (__bf16)f1.z; h[7] = (__bf16)f1.w;
            *(v8bf*)&As[r * KSTR + xc] = h;
        }
        // ---- stage W tile transposed ([n][k]); global reads coalesced in n ----
        {
            v8bf h0, h1;
#pragma unroll
            for (int i = 0; i < 8; ++i)
                h0[i] = (__bf16)Wc[(size_t)(k0 + wk + i) * HID_DIM + wn];
#pragma unroll
            for (int i = 0; i < 8; ++i)
                h1[i] = (__bf16)Wc[(size_t)(k0 + wk + 8 + i) * HID_DIM + wn];
            *(v8bf*)&Bs[wn * KSTR + wk]     = h0;
            *(v8bf*)&Bs[wn * KSTR + wk + 8] = h1;
        }
        __syncthreads();

        // ---- build fragments and issue WMMAs ----
        const int half  = lane >> 4;         // 0: lanes 0-15, 1: lanes 16-31
        const int mrow  = lane & 15;         // A: M row / B: N col
        const int aksel = half * 8;          // A half-lane K select

        v16bf afrag[4];
#pragma unroll
        for (int mt = 0; mt < 4; ++mt) {
            const __bf16* p = &As[(waveM * 64 + mt * 16 + mrow) * KSTR + aksel];
            v8bf lo = *(const v8bf*)(p);       // K = aksel .. aksel+7
            v8bf hi = *(const v8bf*)(p + 16);  // K = aksel+16 .. aksel+23
#pragma unroll
            for (int i = 0; i < 8; ++i) { afrag[mt][i] = lo[i]; afrag[mt][i + 8] = hi[i]; }
        }

        v16bf bfrag[2];
#pragma unroll
        for (int nt = 0; nt < 2; ++nt) {
            const __bf16* p = &Bs[(waveN * 32 + nt * 16 + mrow) * KSTR + half * 8];
            v8bf lo = *(const v8bf*)(p);       // K = half*8 .. +7
            v8bf hi = *(const v8bf*)(p + 16);  // K = half*8+16 .. +23
#pragma unroll
            for (int i = 0; i < 8; ++i) { bfrag[nt][i] = lo[i]; bfrag[nt][i + 8] = hi[i]; }
        }

#pragma unroll
        for (int mt = 0; mt < 4; ++mt)
#pragma unroll
            for (int nt = 0; nt < 2; ++nt)
                acc[mt][nt] = __builtin_amdgcn_wmma_f32_16x16x32_bf16(
                    /*neg_a=*/false, afrag[mt],
                    /*neg_b=*/false, bfrag[nt],
                    /*c_mod=*/(short)0, acc[mt][nt],
                    /*reuse_a=*/false, /*reuse_b=*/false);

        __syncthreads();
    }

    // ---- epilogue: add bias, store f32 ----
    // C/D layout: VGPR v -> M = v + 8*(lane>=16); N = lane & 15.
    const int col  = lane & 15;
    const int rsel = (lane >> 4) * 8;
#pragma unroll
    for (int mt = 0; mt < 4; ++mt) {
#pragma unroll
        for (int nt = 0; nt < 2; ++nt) {
            const int n  = gN + waveN * 32 + nt * 16 + col;
            const float bv = bias[(size_t)cat * HID_DIM + n];
            float* op = out + ((size_t)b * SEQ + gM + waveM * 64 + mt * 16) * HID_DIM + n;
#pragma unroll
            for (int v = 0; v < 8; ++v)
                op[(size_t)(v + rsel) * HID_DIM] = acc[mt][nt][v] + bv;
        }
    }
}

extern "C" void kernel_launch(void* const* d_in, const int* in_sizes, int n_in,
                              void* d_out, int out_size, void* d_ws, size_t ws_size,
                              hipStream_t stream) {
    (void)in_sizes; (void)n_in; (void)out_size; (void)d_ws; (void)ws_size;
    const float* x    = (const float*)d_in[0];
    const int*   cid  = (const int*)  d_in[1];
    const float* W    = (const float*)d_in[2];
    const float* bias = (const float*)d_in[3];
    float*       out  = (float*)d_out;

    dim3 grid(HID_DIM / BN, SEQ / BM, BATCH);   // 32 x 4 x 64
    dim3 block(256, 1, 1);
    cat_linear_wmma_kernel<<<grid, block, 0, stream>>>(x, cid, W, bias, out);
}
